// HCFF_26456998544015
// MI455X (gfx1250) — matrix operand order repaired and checked by reading, not compiled, not run on hardware
//
#include <hip/hip_runtime.h>
#include <hip/hip_bf16.h>
#include <cstdint>

// ---------------------------------------------------------------------------
// CDNA5 (gfx1250) wave32 WMMA helpers: v_wmma_f32_16x16x32_bf16
// ---------------------------------------------------------------------------
typedef __attribute__((ext_vector_type(16))) __bf16 v16bf;
typedef __attribute__((ext_vector_type(8)))  __bf16 v8bf;
typedef __attribute__((ext_vector_type(8)))  float  v8f;

__device__ __forceinline__ v8f vzero8() {
    v8f z;
#pragma unroll
    for (int i = 0; i < 8; ++i) z[i] = 0.0f;
    return z;
}

__device__ __forceinline__ v8f wmma_bf16(v16bf a, v16bf b, v8f c) {
    // (neg_a, A, neg_b, B, c_mod, C, reuse_a, reuse_b)
    return __builtin_amdgcn_wmma_f32_16x16x32_bf16(false, a, false, b, (short)0, c,
                                                   false, false);
}

__device__ __forceinline__ v16bf cat8(v8bf lo, v8bf hi) {
    return __builtin_shufflevector(lo, hi, 0, 1, 2, 3, 4, 5, 6, 7, 8, 9, 10, 11, 12,
                                   13, 14, 15);
}

// A-fragment (16x32 bf16) from LDS row-major [16][S]: lane's data is two
// contiguous 8-element runs -> two aligned ds_load_b128.
// lanes 0-15: row M=lane, K = {0..15}; lanes 16-31: row M=lane-16,
// K = {16..31} (all + kbase).
template <int S>
__device__ __forceinline__ v16bf load_a_lds(const __bf16* buf, int lane, int kbase) {
    int m  = lane & 15;
    int kb = (lane & 16) ? 16 : 0;
    const __bf16* p = buf + m * S + kbase + kb;
    v8bf lo = *(const v8bf*)(p);
    v8bf hi = *(const v8bf*)(p + 8);
    return cat8(lo, hi);
}

// B-fragment (32x16 bf16) from a fragment-packed LDS tile [32 lanes][16]:
// one aligned 32-byte run per lane -> two ds_load_b128.
// Packing rule: frag[l][e] = B[((l&16)?16:0) + e][(l&15)].
__device__ __forceinline__ v16bf load_b_frag(const __bf16* tile, int lane) {
    const __bf16* p = tile + lane * 16;
    v8bf lo = *(const v8bf*)(p);
    v8bf hi = *(const v8bf*)(p + 8);
    return cat8(lo, hi);
}

__device__ __forceinline__ void lds_fence() {
    asm volatile("s_wait_dscnt 0x0" ::: "memory");
}

// ---------------------------------------------------------------------------
// Kernel 1: brute-force KNN (pq (B,3,N) queries vs ps (B,3,M) support)
// ---------------------------------------------------------------------------
template <int KN>
__global__ void knn_kernel(const float* __restrict__ pq, const float* __restrict__ ps,
                           int N, int M, int* __restrict__ idx) {
    __shared__ float sx[256], sy[256], sz[256];
    const int b    = blockIdx.y;
    const int n    = blockIdx.x * 256 + threadIdx.x;
    const bool vld = (n < N);
    const int nc   = vld ? n : 0;
    const float px = pq[(size_t)(b * 3 + 0) * N + nc];
    const float py = pq[(size_t)(b * 3 + 1) * N + nc];
    const float pz = pq[(size_t)(b * 3 + 2) * N + nc];
    float bd[KN];
    int   bi[KN];
#pragma unroll
    for (int t = 0; t < KN; ++t) { bd[t] = 3.4e38f; bi[t] = 0; }

    for (int m0 = 0; m0 < M; m0 += 256) {
        int ml = m0 + threadIdx.x;
        int mc = (ml < M) ? ml : 0;
        sx[threadIdx.x] = ps[(size_t)(b * 3 + 0) * M + mc];
        sy[threadIdx.x] = ps[(size_t)(b * 3 + 1) * M + mc];
        sz[threadIdx.x] = ps[(size_t)(b * 3 + 2) * M + mc];
        __syncthreads();
        int lim = (M - m0) < 256 ? (M - m0) : 256;
        for (int t = 0; t < lim; ++t) {
            float dx = px - sx[t], dy = py - sy[t], dz = pz - sz[t];
            float d  = dx * dx + dy * dy + dz * dz;
            if (d < bd[KN - 1]) {
                bd[KN - 1] = d;
                bi[KN - 1] = m0 + t;
#pragma unroll
                for (int p = KN - 1; p > 0; --p) {
                    if (bd[p] < bd[p - 1]) {
                        float td = bd[p]; bd[p] = bd[p - 1]; bd[p - 1] = td;
                        int   ti = bi[p]; bi[p] = bi[p - 1]; bi[p - 1] = ti;
                    }
                }
            }
        }
        __syncthreads();
    }
    if (vld) {
#pragma unroll
        for (int t = 0; t < KN; ++t) idx[((size_t)b * N + n) * KN + t] = bi[t];
    }
}

// ---------------------------------------------------------------------------
// Kernel 2: q/k/v projections. out(64,N) = W(64,C) x F(C,N) + bias.
// Block = 256 (8 waves), block tile M=64 x N=128, K-chunks of 32.
// Weights row-major in LDS (A-operand, b128 runs); activations staged
// fragment-packed (B-operand, b128 runs). grid = (N/128, B, 3).
// ---------------------------------------------------------------------------
__global__ void __launch_bounds__(256) qkv_kernel(
    const float* __restrict__ fq, const float* __restrict__ fs,
    const float* __restrict__ wq, const float* __restrict__ bq,
    const float* __restrict__ wk, const float* __restrict__ bk,
    const float* __restrict__ wv, const float* __restrict__ bv,
    int C, int N,
    float* __restrict__ qo, float* __restrict__ ko, float* __restrict__ vo) {
    constexpr int WS = 40;
    __shared__ __bf16 sW[64][WS] __attribute__((aligned(32)));   // [m][k]
    __shared__ __bf16 sFf[8][512] __attribute__((aligned(32)));  // frag-packed

    const int tid  = threadIdx.x;
    const int lane = tid & 31;
    const int wv_  = tid >> 5;
    const int nb0  = blockIdx.x * 128;
    const int b    = blockIdx.y;
    const int z    = blockIdx.z;
    const float* W    = (z == 0) ? wq : (z == 1) ? wk : wv;
    const float* bias = (z == 0) ? bq : (z == 1) ? bk : bv;
    const float* F    = ((z == 0) ? fq : fs) + (size_t)b * C * N;
    float*       O    = ((z == 0) ? qo : (z == 1) ? ko : vo) + (size_t)b * 64 * N;

    v8f acc[4];
#pragma unroll
    for (int mt = 0; mt < 4; ++mt) acc[mt] = vzero8();

    for (int kc = 0; kc < C; kc += 32) {
        // stage weight chunk 64x32 row-major (A-operand)
#pragma unroll
        for (int i = tid; i < 64 * 32; i += 256) {
            int m = i >> 5, k = i & 31;
            sW[m][k] = (__bf16)W[m * C + kc + k];
        }
        // stage activation chunk 32x128 fragment-packed (coalesced float4 in)
#pragma unroll
        for (int i = tid; i < 32 * 32; i += 256) {
            int k = i >> 5, c4 = (i & 31) << 2;
            const float4 f = *(const float4*)&F[(size_t)(kc + k) * N + nb0 + c4];
            float vals[4] = {f.x, f.y, f.z, f.w};
            int lbase = (k & 16) ? 16 : 0;
            int e     = k & 15;
#pragma unroll
            for (int cc = 0; cc < 4; ++cc) {
                int col = c4 + cc;
                sFf[col >> 4][(lbase + (col & 15)) * 16 + e] = (__bf16)vals[cc];
            }
        }
        if (kc + 32 < C) {  // prefetch next chunk (global_prefetch_b8)
            __builtin_prefetch(&F[(size_t)(kc + 32 + (tid >> 3)) * N + nb0 + ((tid & 7) << 4)], 0, 0);
        }
        __syncthreads();
        v16bf bf = load_b_frag(sFf[wv_], lane);
#pragma unroll
        for (int mt = 0; mt < 4; ++mt) {
            v16bf af = load_a_lds<WS>(&sW[mt * 16][0], lane, 0);
            acc[mt]  = wmma_bf16(af, bf, acc[mt]);
        }
        __syncthreads();
    }
    const int col = nb0 + wv_ * 16 + (lane & 15);
#pragma unroll
    for (int mt = 0; mt < 4; ++mt) {
#pragma unroll
        for (int r = 0; r < 8; ++r) {
            int row = mt * 16 + ((lane & 16) ? (8 + r) : r);
            O[row * N + col] = acc[mt][r] + bias[row];
        }
    }
}

// ---------------------------------------------------------------------------
// Kernel 3: fused VFA attention. One wave = one 16-row tile
// (QPT queries x padded K neighbors). Weights fragment-packed in LDS.
// ---------------------------------------------------------------------------
template <int KN, int QPT>
__global__ void __launch_bounds__(256) attn_kernel(
    const float* __restrict__ pq, const float* __restrict__ ps,
    const float* __restrict__ qbuf, const float* __restrict__ kbuf,
    const float* __restrict__ vbuf, const int* __restrict__ idx,
    const float* __restrict__ pw1, const float* __restrict__ pb1,
    const float* __restrict__ pg,  const float* __restrict__ pbe,
    const float* __restrict__ pm,  const float* __restrict__ pv,
    const float* __restrict__ pw2, const float* __restrict__ pb2,
    const float* __restrict__ aw1, const float* __restrict__ ab1,
    const float* __restrict__ ag,  const float* __restrict__ abe,
    const float* __restrict__ am,  const float* __restrict__ av,
    const float* __restrict__ aw2, const float* __restrict__ ab2,
    int N, float* __restrict__ agg) {
    constexpr int XS = 72, HS = 40;
    // fragment-packed weight tiles (32K x 16cols each, [lane][16] layout)
    __shared__ __bf16 s_pw2f[8][512]  __attribute__((aligned(32)));  // kc*4+nt
    __shared__ __bf16 s_aw1f[32][512] __attribute__((aligned(32)));  // kc*16+ct
    __shared__ __bf16 s_aw2f[32][512] __attribute__((aligned(32)));  // hc*4+nt
    __shared__ float  s_pw1[192];  // (64,3) row-major
    __shared__ float  s_sc1[64], s_sh1[64], s_pb2b[64], s_ab2b[64];
    __shared__ float  s_sc2[256], s_sh2[256];
    // per-wave staging
    __shared__ __bf16 s_x[8][16][XS] __attribute__((aligned(32)));
    __shared__ __bf16 s_h[8][16][HS] __attribute__((aligned(32)));
    __shared__ int    s_qo[8][16], s_ko[8][16], s_mv[8][16];
    __shared__ float  s_pos[8][16][3];

    // ---- stage weights (fragment-packed; inner 16 reads are contiguous) ----
    for (int i = threadIdx.x; i < 8 * 512; i += 256) {
        int tile = i >> 9, r = i & 511, l = r >> 4, e = r & 15;
        int kc = tile >> 2, nt = tile & 3;
        int k = kc * 32 + ((l & 16) ? 16 : 0) + e;
        int col = nt * 16 + (l & 15);
        s_pw2f[tile][r] = (__bf16)pw2[col * 64 + k];  // B[k=c][col=o] = pw2[o][c]
    }
    for (int i = threadIdx.x; i < 32 * 512; i += 256) {
        int tile = i >> 9, r = i & 511, l = r >> 4, e = r & 15;
        int kc = tile >> 4, ct = tile & 15;
        int k = kc * 32 + ((l & 16) ? 16 : 0) + e;
        int col = ct * 16 + (l & 15);
        s_aw1f[tile][r] = (__bf16)aw1[col * 64 + k];  // B[k=c][col=h] = aw1[h][c]
    }
    for (int i = threadIdx.x; i < 32 * 512; i += 256) {
        int tile = i >> 9, r = i & 511, l = r >> 4, e = r & 15;
        int hc = tile >> 2, nt = tile & 3;
        int k = hc * 32 + ((l & 16) ? 16 : 0) + e;
        int col = nt * 16 + (l & 15);
        s_aw2f[tile][r] = (__bf16)aw2[col * 256 + k];  // B[k=h][col=o] = aw2[o][h]
    }
    for (int i = threadIdx.x; i < 192; i += 256) s_pw1[i] = pw1[i];
    for (int i = threadIdx.x; i < 64; i += 256) {
        float sc  = pg[i] * rsqrtf(pv[i] + 1e-5f);
        s_sc1[i]  = sc;
        s_sh1[i]  = (pb1[i] - pm[i]) * sc + pbe[i];  // fold conv bias + BN
        s_pb2b[i] = pb2[i];
        s_ab2b[i] = ab2[i];
    }
    for (int i = threadIdx.x; i < 256; i += 256) {
        float sc = ag[i] * rsqrtf(av[i] + 1e-5f);
        s_sc2[i] = sc;
        s_sh2[i] = (ab1[i] - am[i]) * sc + abe[i];
    }
    __syncthreads();

    const int lane = threadIdx.x & 31;
    const int wv_  = threadIdx.x >> 5;
    __bf16 (*xbuf)[XS] = s_x[wv_];
    __bf16 (*hbuf)[HS] = s_h[wv_];
    int* mqo = s_qo[wv_];
    int* mko = s_ko[wv_];
    int* mv  = s_mv[wv_];

    const int totalq = 2 * N;  // B = 2
    const int tiles  = (totalq + QPT - 1) / QPT;
    const int nwaves = gridDim.x * 8;
    const int wid    = blockIdx.x * 8 + wv_;
    constexpr int RPQ = 16 / QPT;

    for (int t = wid; t < tiles; t += nwaves) {
        // ---- per-row metadata (lanes 0..15, one row each) ----
        if (lane < 16) {
            int r   = lane;
            int sub = r / RPQ;
            int kk  = r % RPQ;
            int qi  = t * QPT + sub;
            bool qv = (qi < totalq);
            int qc  = qv ? qi : (totalq - 1);
            int bb  = qc / N, nn = qc % N;
            bool valid = qv && (kk < KN);
            int j = idx[(size_t)qc * KN + ((kk < KN) ? kk : 0)];
            // fused int32 offsets: element [c][n] of (B,64,N) = bb*64*N + c*N + n
            mqo[r] = bb * 64 * N + nn;
            mko[r] = bb * 64 * N + j;
            mv[r]  = valid ? 1 : 0;
            s_pos[wv_][r][0] = pq[(bb * 3 + 0) * N + nn] - ps[(bb * 3 + 0) * N + j];
            s_pos[wv_][r][1] = pq[(bb * 3 + 1) * N + nn] - ps[(bb * 3 + 1) * N + j];
            s_pos[wv_][r][2] = pq[(bb * 3 + 2) * N + nn] - ps[(bb * 3 + 2) * N + j];
        }
        lds_fence();

        int qo8[8], ko8[8], vv8[8];
#pragma unroll
        for (int r = 0; r < 8; ++r) {
            int row = (lane & 16) ? (r + 8) : r;
            qo8[r] = mqo[row];
            ko8[r] = mko[row];
            vv8[r] = mv[row];
        }

        // ---- pe1 = relu(bn(pw1 * pos_rel)) into xbuf (bf16 A-operand) ----
        {
            int m  = lane & 15;
            int cb = (lane & 16) ? 32 : 0;
            float px = s_pos[wv_][m][0], py = s_pos[wv_][m][1], pz = s_pos[wv_][m][2];
#pragma unroll
            for (int c0 = 0; c0 < 32; ++c0) {
                int c   = cb + c0;
                float v = s_pw1[c * 3 + 0] * px + s_pw1[c * 3 + 1] * py + s_pw1[c * 3 + 2] * pz;
                v       = fmaxf(v * s_sc1[c] + s_sh1[c], 0.0f);
                xbuf[m][c] = (__bf16)v;
            }
        }
        lds_fence();

        // ---- pe = pw2 * pe1 + pb2  (8 WMMAs) ----
        v8f pe_acc[4];
#pragma unroll
        for (int nt = 0; nt < 4; ++nt) pe_acc[nt] = vzero8();
#pragma unroll
        for (int kc = 0; kc < 2; ++kc) {
            v16bf a = load_a_lds<XS>(&xbuf[0][0], lane, kc * 32);
#pragma unroll
            for (int nt = 0; nt < 4; ++nt)
                pe_acc[nt] = wmma_bf16(a, load_b_frag(s_pw2f[kc * 4 + nt], lane), pe_acc[nt]);
        }
#pragma unroll
        for (int nt = 0; nt < 4; ++nt) {
            float bias = s_pb2b[nt * 16 + (lane & 15)];
#pragma unroll
            for (int r = 0; r < 8; ++r) pe_acc[nt][r] += bias;
        }

        // ---- x2 = (q - k_grouped) + pe  -> xbuf (A-operand for aw1) ----
#pragma unroll
        for (int nt = 0; nt < 4; ++nt) {
            int c  = nt * 16 + (lane & 15);
            int cN = c * N;
#pragma unroll
            for (int r = 0; r < 8; ++r) {
                int row  = (lane & 16) ? (r + 8) : r;
                float qv = qbuf[cN + qo8[r]];
                float kv = kbuf[cN + ko8[r]];
                xbuf[row][c] = (__bf16)(qv - kv + pe_acc[nt][r]);
            }
        }
        lds_fence();
        v16bf xa0 = load_a_lds<XS>(&xbuf[0][0], lane, 0);
        v16bf xa1 = load_a_lds<XS>(&xbuf[0][0], lane, 32);

        // ---- attention MLP: h = relu(bn(aw1 x)), logits = aw2 h + ab2 ----
        v8f a_acc[4];
#pragma unroll
        for (int nt = 0; nt < 4; ++nt) {
            float bias = s_ab2b[nt * 16 + (lane & 15)];
#pragma unroll
            for (int r = 0; r < 8; ++r) a_acc[nt][r] = bias;
        }
#pragma unroll
        for (int hc = 0; hc < 8; ++hc) {  // 256 hidden in chunks of 32
            v8f h0 = vzero8(), h1 = vzero8();
            h0 = wmma_bf16(xa0, load_b_frag(s_aw1f[hc * 2], lane), h0);
            h0 = wmma_bf16(xa1, load_b_frag(s_aw1f[16 + hc * 2], lane), h0);
            h1 = wmma_bf16(xa0, load_b_frag(s_aw1f[hc * 2 + 1], lane), h1);
            h1 = wmma_bf16(xa1, load_b_frag(s_aw1f[16 + hc * 2 + 1], lane), h1);
            {
                int h0c = hc * 32 + (lane & 15);
                int h1c = h0c + 16;
                float sc0 = s_sc2[h0c], sh0 = s_sh2[h0c];
                float sc1 = s_sc2[h1c], sh1 = s_sh2[h1c];
#pragma unroll
                for (int r = 0; r < 8; ++r) {
                    int row = (lane & 16) ? (r + 8) : r;
                    hbuf[row][(lane & 15)]      = (__bf16)fmaxf(h0[r] * sc0 + sh0, 0.0f);
                    hbuf[row][16 + (lane & 15)] = (__bf16)fmaxf(h1[r] * sc1 + sh1, 0.0f);
                }
            }
            lds_fence();
            v16bf ah = load_a_lds<HS>(&hbuf[0][0], lane, 0);
#pragma unroll
            for (int nt = 0; nt < 4; ++nt)
                a_acc[nt] = wmma_bf16(ah, load_b_frag(s_aw2f[hc * 4 + nt], lane), a_acc[nt]);
        }

        // ---- per-channel softmax over neighbors; a_acc becomes weights ----
#pragma unroll
        for (int nt = 0; nt < 4; ++nt) {
#pragma unroll
            for (int r = 0; r < 8; ++r)
                if (!vv8[r]) a_acc[nt][r] = -1.0e30f;
            float mx = -1.0e30f;
#pragma unroll
            for (int r = 0; r < 8; ++r) mx = fmaxf(mx, a_acc[nt][r]);
            if (QPT == 1) mx = fmaxf(mx, __shfl_xor(mx, 16, 32));
            float sm = 0.0f;
#pragma unroll
            for (int r = 0; r < 8; ++r) {
                float e      = __expf(a_acc[nt][r] - mx);
                a_acc[nt][r] = e;
                sm += e;
            }
            if (QPT == 1) sm += __shfl_xor(sm, 16, 32);
            float inv = 1.0f / sm;
#pragma unroll
            for (int r = 0; r < 8; ++r) a_acc[nt][r] *= inv;
        }

        // ---- aggregate: agg[c] = sum_k w[k] * (v[c, idx_k] + pe[c,k]) ----
        bool qvalid = (vv8[0] != 0);
#pragma unroll
        for (int nt = 0; nt < 4; ++nt) {
            int c  = nt * 16 + (lane & 15);
            int cN = c * N;
            float s = 0.0f;
#pragma unroll
            for (int r = 0; r < 8; ++r) {
                float vv = vbuf[cN + ko8[r]];
                s += a_acc[nt][r] * (vv + pe_acc[nt][r]);
            }
            if (QPT == 1) {
                s += __shfl_xor(s, 16, 32);
                if (lane < 16 && qvalid) agg[cN + qo8[0]] = s;
            } else {
                if (qvalid) agg[cN + qo8[0]] = s;
            }
        }
    }
}

// ---------------------------------------------------------------------------
// Kernel 4: out = we(C,64) @ agg(64,N) + be + fq (residual projection).
// Block = 256 (8 waves), block tile M=64 x N=128, K=64 in 2 chunks.
// ---------------------------------------------------------------------------
__global__ void __launch_bounds__(256) proj_kernel(
    const float* __restrict__ agg, const float* __restrict__ we,
    const float* __restrict__ be, const float* __restrict__ fq,
    int C, int N, float* __restrict__ out) {
    constexpr int WS = 40;
    __shared__ __bf16 sW[64][WS] __attribute__((aligned(32)));
    __shared__ __bf16 sBf[8][512] __attribute__((aligned(32)));

    const int tid  = threadIdx.x;
    const int lane = tid & 31;
    const int wv_  = tid >> 5;
    const int nb0  = blockIdx.x * 128;
    const int cb   = blockIdx.y * 64;
    const int b    = blockIdx.z;
    const float* Bm = agg + (size_t)b * 64 * N;

    v8f acc[4];
#pragma unroll
    for (int mt = 0; mt < 4; ++mt) acc[mt] = vzero8();

#pragma unroll
    for (int kc = 0; kc < 64; kc += 32) {
#pragma unroll
        for (int i = tid; i < 64 * 32; i += 256) {
            int m = i >> 5, k = i & 31;
            sW[m][k] = (__bf16)we[(cb + m) * 64 + kc + k];
        }
#pragma unroll
        for (int i = tid; i < 32 * 32; i += 256) {
            int k = i >> 5, c4 = (i & 31) << 2;
            const float4 f = *(const float4*)&Bm[(size_t)(kc + k) * N + nb0 + c4];
            float vals[4] = {f.x, f.y, f.z, f.w};
            int lbase = (k & 16) ? 16 : 0;
            int e     = k & 15;
#pragma unroll
            for (int cc = 0; cc < 4; ++cc) {
                int col = c4 + cc;
                sBf[col >> 4][(lbase + (col & 15)) * 16 + e] = (__bf16)vals[cc];
            }
        }
        __syncthreads();
        v16bf bf = load_b_frag(sBf[wv_], lane);
#pragma unroll
        for (int mt = 0; mt < 4; ++mt) {
            v16bf af = load_a_lds<WS>(&sW[mt * 16][0], lane, 0);
            acc[mt]  = wmma_bf16(af, bf, acc[mt]);
        }
        __syncthreads();
    }
    const int n = nb0 + wv_ * 16 + (lane & 15);
#pragma unroll
    for (int mt = 0; mt < 4; ++mt) {
#pragma unroll
        for (int r = 0; r < 8; ++r) {
            int c      = cb + mt * 16 + ((lane & 16) ? (r + 8) : r);
            size_t off = ((size_t)b * C + c) * N + n;
            out[off]   = acc[mt][r] + be[c] + fq[off];
        }
    }
}

// ---------------------------------------------------------------------------
// Host launcher
// ---------------------------------------------------------------------------
extern "C" void kernel_launch(void* const* d_in, const int* in_sizes, int n_in,
                              void* d_out, int out_size, void* d_ws, size_t ws_size,
                              hipStream_t stream) {
    (void)in_sizes; (void)n_in; (void)out_size; (void)ws_size;
    const int Bb = 2;
    const int CH[3]  = {512, 256, 128};
    const int NSZ[3] = {2048, 4096, 8192};
    const int KNS[3] = {16, 12, 8};

    // workspace carve-out (max over scales): idx + q/k/v/agg  (~16.5 MB)
    char*  w    = (char*)d_ws;
    int*   idxb = (int*)w;   w += (size_t)131072 * sizeof(int);
    float* qb   = (float*)w; w += (size_t)1048576 * sizeof(float);
    float* kb   = (float*)w; w += (size_t)1048576 * sizeof(float);
    float* vb   = (float*)w; w += (size_t)1048576 * sizeof(float);
    float* aggb = (float*)w; w += (size_t)1048576 * sizeof(float);

    size_t outoff = 0;
    for (int s = 2; s >= 0; --s) {  // fine -> coarse, matching reference order
        const int C = CH[s], N = NSZ[s], K = KNS[s];
        const float* pp = (const float*)d_in[4 * s + 0];
        const float* pf = (const float*)d_in[4 * s + 1];
        const float* gp = (const float*)d_in[4 * s + 2];
        const float* gf = (const float*)d_in[4 * s + 3];
        const float* P[24];
        for (int j = 0; j < 24; ++j) P[j] = (const float*)d_in[12 + 24 * s + j];

        // 1) KNN: queries pp vs support gp
        dim3 gk((N + 255) / 256, Bb);
        if (K == 16)      knn_kernel<16><<<gk, 256, 0, stream>>>(pp, gp, N, N, idxb);
        else if (K == 12) knn_kernel<12><<<gk, 256, 0, stream>>>(pp, gp, N, N, idxb);
        else              knn_kernel<8><<<gk, 256, 0, stream>>>(pp, gp, N, N, idxb);

        // 2) q/k/v projections (q from pf; k,v from gf)
        dim3 gq(N / 128, Bb, 3);
        qkv_kernel<<<gq, 256, 0, stream>>>(pf, gf, P[0], P[1], P[2], P[3], P[4], P[5],
                                           C, N, qb, kb, vb);

        // 3) fused attention -> agg (B,64,N)
        dim3 ga(512);
        if (K == 16)
            attn_kernel<16, 1><<<ga, 256, 0, stream>>>(
                pp, gp, qb, kb, vb, idxb, P[6], P[7], P[8], P[9], P[10], P[11], P[12],
                P[13], P[14], P[15], P[16], P[17], P[18], P[19], P[20], P[21], N, aggb);
        else if (K == 12)
            attn_kernel<12, 1><<<ga, 256, 0, stream>>>(
                pp, gp, qb, kb, vb, idxb, P[6], P[7], P[8], P[9], P[10], P[11], P[12],
                P[13], P[14], P[15], P[16], P[17], P[18], P[19], P[20], P[21], N, aggb);
        else
            attn_kernel<8, 2><<<ga, 256, 0, stream>>>(
                pp, gp, qb, kb, vb, idxb, P[6], P[7], P[8], P[9], P[10], P[11], P[12],
                P[13], P[14], P[15], P[16], P[17], P[18], P[19], P[20], P[21], N, aggb);

        // 4) residual output projection
        dim3 gpj(N / 128, C / 64, Bb);
        proj_kernel<<<gpj, 256, 0, stream>>>(aggb, P[22], P[23], pf, C, N,
                                             (float*)d_out + outoff);
        outoff += (size_t)Bb * C * N;
    }
}
